// UMGMQuantizer_44573170597906
// MI455X (gfx1250) — matrix operand order repaired and checked by reading, not compiled, not run on hardware
//
#include <hip/hip_runtime.h>
#include <hip/hip_bf16.h>

// Problem constants (from reference)
#define NB 4
#define CC 256
#define HWX 1024          // H*W
#define MMQ 4
#define KKQ 2048
#define DDQ 64
#define EPS_T 3.814697265625e-06f           // 2^-18 temperature lower bound
#define F32_EPS 1.1920928955078125e-07f     // jnp.finfo(f32).eps
#define INV_SQRT_K 0.02209708691207961f     // 1/sqrt(2048)

typedef __attribute__((ext_vector_type(2)))  float  v2f;
typedef __attribute__((ext_vector_type(8)))  float  v8f;
typedef __attribute__((ext_vector_type(16))) __bf16 v16bf;

// ---------------------------------------------------------------------------
// CDNA5 async global->LDS copy (8 bytes/lane), tracked by ASYNCcnt.
// vdst = LDS byte address (flat-truncated), vaddr = 64-bit global address.
// ---------------------------------------------------------------------------
__device__ __forceinline__ void async_copy_b64(unsigned lds_addr, const void* gaddr)
{
    asm volatile("global_load_async_to_lds_b64 %0, %1, off"
                 :: "v"(lds_addr), "v"((unsigned long long)(uintptr_t)gaddr)
                 : "memory");
}
__device__ __forceinline__ void wait_asynccnt0()
{
    asm volatile("s_wait_asynccnt 0x0" ::: "memory");
}

// ---------------------------------------------------------------------------
// conv1x1 as GEMM: out[n,o,p] = sum_c W1[o,c]*A1[n,c,p] (+ W2[o,c]*A2[n,c,p])
// One wave32 computes one 16(o) x 16(pos) tile with V_WMMA_F32_16X16X4_F32.
// ---------------------------------------------------------------------------
__global__ __launch_bounds__(256) void gemm1x1_wmma(
    float* __restrict__ out,
    const float* __restrict__ W1, const float* __restrict__ A1,
    const float* __restrict__ W2, const float* __restrict__ A2)
{
    const int lane = threadIdx.x & 31;
    const int wv   = threadIdx.x >> 5;
    const int tile = blockIdx.x * 8 + wv;        // 4096 tiles total
    const int pt = tile & 63;                    // 64 pos tiles per image
    const int ot = (tile >> 6) & 15;             // 16 output-channel tiles
    const int n  = tile >> 10;                   // 4 images
    const int pb = pt * 16, ob = ot * 16;
    const int hf  = lane >> 4;                   // half-wave id
    const int l16 = lane & 15;

    v8f acc = {};
    {
        const float* a  = A1 + (size_t)n * CC * HWX;
        const float* wr = W1 + (size_t)(ob + l16) * CC;
        __builtin_prefetch(a + pb, 0, 0);        // global_prefetch_b8
        for (int c = 0; c < CC; c += 4) {
            v2f av, bv;
            // A frag (32-bit 16x4): lanes0-15 K=0,1 ; lanes16-31 K=2,3
            av.x = wr[c + 2 * hf];
            av.y = wr[c + 2 * hf + 1];
            // B frag (32-bit 4x16): VGPRv: lanes0-15 K=v ; lanes16-31 K=v+2
            bv.x = a[(size_t)(c + 2 * hf) * HWX + pb + l16];
            bv.y = a[(size_t)(c + 2 * hf + 1) * HWX + pb + l16];
            acc = __builtin_amdgcn_wmma_f32_16x16x4_f32(
                      false, av, false, bv, (short)0, acc, false, false);
        }
    }
    if (W2 != nullptr) {                          // fused second GEMM (decode sum)
        const float* a  = A2 + (size_t)n * CC * HWX;
        const float* wr = W2 + (size_t)(ob + l16) * CC;
        for (int c = 0; c < CC; c += 4) {
            v2f av, bv;
            av.x = wr[c + 2 * hf];
            av.y = wr[c + 2 * hf + 1];
            bv.x = a[(size_t)(c + 2 * hf) * HWX + pb + l16];
            bv.y = a[(size_t)(c + 2 * hf + 1) * HWX + pb + l16];
            acc = __builtin_amdgcn_wmma_f32_16x16x4_f32(
                      false, av, false, bv, (short)0, acc, false, false);
        }
    }
    float* o = out + (size_t)n * CC * HWX;
#pragma unroll
    for (int v = 0; v < 8; ++v)
        o[(size_t)(ob + v + 8 * hf) * HWX + pb + l16] = acc[v];
}

// ---------------------------------------------------------------------------
// Codebook prep: f32 -> bf16 copy + |c|^2 table. One thread per (m,k) row.
// ---------------------------------------------------------------------------
__global__ __launch_bounds__(256) void prep_codebook(
    const float* __restrict__ cb, __bf16* __restrict__ cbb, float* __restrict__ c2)
{
    const int i = blockIdx.x * blockDim.x + threadIdx.x;   // over M*K = 8192
    if (i >= MMQ * KKQ) return;
    const float* src = cb + (size_t)i * DDQ;
    __bf16* dst = cbb + (size_t)i * DDQ;
    float s = 0.f;
#pragma unroll
    for (int d = 0; d < DDQ; ++d) {
        const float v = src[d];
        s += v * v;
        dst[d] = (__bf16)v;
    }
    c2[i] = s;
}

// ---------------------------------------------------------------------------
// Fused quantize+dequantize.
//   score_k = (2*inter_k - |c_k|^2) * max(temp,eps)/sqrt(K) + gumbel(u_k)
// (|x|^2 term dropped: constant per position -> argmax invariant.)
// One wave owns 16 positions, loops over all 128 k-tiles keeping a running
// argmax entirely in registers; logits are never written to memory.
// Codebook k-tiles (16x64 bf16 = 2KB) are double-buffered in LDS via
// GLOBAL_LOAD_ASYNC_TO_LDS_B64 (ASYNCcnt): tile i+1 streams in while the
// WMMAs consume tile i. The 268MB gumbel-noise stream is read with
// non-temporal hints so it does not thrash L2.
// ---------------------------------------------------------------------------
__global__ __launch_bounds__(256) void quantize_wmma(
    float* __restrict__ quant,       // [N,C,HW] (writes the m-slice)
    const float* __restrict__ q,     // [N,C,HW]
    const float* __restrict__ cbf,   // [M,K,D] f32 (exact dequant values)
    const __bf16* __restrict__ cbb,  // [M,K,D] bf16
    const float* __restrict__ c2,    // [M,K]
    const float* __restrict__ temp,  // [M]
    const float* __restrict__ u)     // [N,M,K,HW]
{
    __shared__ __bf16 ldsA[2][16 * DDQ];         // double-buffered k-tile (2x2KB)
    const int lane = threadIdx.x & 31;
    const int wv   = threadIdx.x >> 5;
    const int pg = blockIdx.x & 7;               // 8 pos-groups
    const int m  = (blockIdx.x >> 3) & 3;
    const int n  = blockIdx.x >> 5;
    const int pb  = (pg * 8 + wv) * 16;
    const int hf  = lane >> 4;
    const int l16 = lane & 15;
    const int pos = pb + l16;

    // B fragments (query vector for this position), loaded once, reused 128x.
    // bf16 32x16 B: elem e<8 holds K=8*hf+e, elem e>=8 holds K=16+8*hf+(e-8)
    const float* qn = q + (size_t)n * CC * HWX + (size_t)m * DDQ * HWX;
    v16bf blo, bhi;
#pragma unroll
    for (int e = 0; e < 16; ++e) {
        blo[e] = (__bf16)qn[(size_t)(8 * hf + e + (e >= 8 ? 8 : 0)) * HWX + pos];
        bhi[e] = (__bf16)qn[(size_t)(32 + 8 * hf + e + (e >= 8 ? 8 : 0)) * HWX + pos];
    }

    const float t    = fmaxf(temp[m], EPS_T) * INV_SQRT_K;
    const float* un  = u + (size_t)(n * MMQ + m) * KKQ * HWX;
    const float* c2m = c2 + (size_t)m * KKQ;
    const __bf16* cbm = cbb + (size_t)m * KKQ * DDQ;

    // per-thread 8-byte slice of the cooperative 2KB tile copy
    const int eoff = threadIdx.x << 2;           // element offset (4 bf16 = 8B)
    const unsigned lds0 = (unsigned)(uintptr_t)(&ldsA[0][0] + eoff);
    const unsigned lds1 = (unsigned)(uintptr_t)(&ldsA[1][0] + eoff);

    float best = -__builtin_inff();
    int   bk   = 0;

    // prologue: async-fill buffer 0 with k-tile 0
    async_copy_b64(lds0, cbm + eoff);
    wait_asynccnt0();
    __syncthreads();

    for (int kb = 0; kb < KKQ; kb += 16) {
        const int cur = (kb >> 4) & 1;
        // prefetch next k-tile into the other buffer (overlaps with compute)
        if (kb + 16 < KKQ)
            async_copy_b64(cur ? lds0 : lds1,
                           cbm + (size_t)(kb + 16) * DDQ + eoff);

        // A fragments from LDS (bf16 16x32): row = k_local = l16,
        // contiguous 8-elem chunks at d = 8*hf and 16+8*hf (+32 for hi).
        const __bf16* row = &ldsA[cur][0] + l16 * DDQ;
        v16bf alo, ahi;
#pragma unroll
        for (int e = 0; e < 8; ++e) {
            alo[e]     = row[8 * hf + e];
            alo[e + 8] = row[16 + 8 * hf + e];
            ahi[e]     = row[32 + 8 * hf + e];
            ahi[e + 8] = row[48 + 8 * hf + e];
        }

        v8f acc = {};
        acc = __builtin_amdgcn_wmma_f32_16x16x32_bf16(
                  false, alo, false, blo, (short)0, acc, false, false);
        acc = __builtin_amdgcn_wmma_f32_16x16x32_bf16(
                  false, ahi, false, bhi, (short)0, acc, false, false);

        // C layout: lane holds k_local = v + 8*hf at column pos = l16
#pragma unroll
        for (int v = 0; v < 8; ++v) {
            const int k = kb + v + 8 * hf;
            float uu = __builtin_nontemporal_load(un + (size_t)k * HWX + pos);
            uu = fminf(fmaxf(uu, F32_EPS), 1.0f - F32_EPS);
            const float g  = -__logf(-__logf(uu));           // gumbel noise
            const float sc = fmaf(2.0f * acc[v] - c2m[k], t, g);
            if (sc > best) { best = sc; bk = k; }            // first-max wins
        }

        wait_asynccnt0();      // my prefetch slice landed in LDS
        __syncthreads();       // everyone's slice landed; reads of cur done
    }

    // combine the two half-waves (lane ^ 16 covers the other 8 k's per tile)
    const float bo = __shfl_xor(best, 16, 32);
    const int   ko = __shfl_xor(bk, 16, 32);
    if (bo > best || (bo == best && ko < bk)) { best = bo; bk = ko; }

    // dequantize: exact f32 codebook row of winner; each half writes 32 d's
    const float* cw = cbf + ((size_t)m * KKQ + bk) * DDQ;
    float* qo = quant + (size_t)n * CC * HWX + (size_t)m * DDQ * HWX;
#pragma unroll
    for (int d = 0; d < 32; ++d)
        qo[(size_t)(32 * hf + d) * HWX + pos] = cw[32 * hf + d];
}

// ---------------------------------------------------------------------------
extern "C" void kernel_launch(void* const* d_in, const int* in_sizes, int n_in,
                              void* d_out, int out_size, void* d_ws, size_t ws_size,
                              hipStream_t stream)
{
    (void)in_sizes; (void)n_in; (void)out_size; (void)ws_size;
    const float* x    = (const float*)d_in[0];
    const float* u1   = (const float*)d_in[1];
    const float* u2   = (const float*)d_in[2];
    const float* cb1  = (const float*)d_in[3];
    const float* cb2  = (const float*)d_in[4];
    const float* tmp1 = (const float*)d_in[5];
    const float* tmp2 = (const float*)d_in[6];
    const float* Wstage1 = (const float*)d_in[7];
    const float* Wstage2 = (const float*)d_in[8];
    const float* Wq1     = (const float*)d_in[9];
    const float* Wq2     = (const float*)d_in[10];
    const float* Wlat1   = (const float*)d_in[11];
    const float* Wdeq1   = (const float*)d_in[12];
    const float* Wdeq2   = (const float*)d_in[13];
    const float* Wside1  = (const float*)d_in[14];
    const float* Wrest1  = (const float*)d_in[15];
    const float* Wrest2  = (const float*)d_in[16];
    float* out = (float*)d_out;

    char* ws = (char*)d_ws;
    const size_t SZ = (size_t)NB * CC * HWX * sizeof(float);      // 4 MB
    float*  bufZ   = (float*)(ws);                                // scratch
    float*  bufA   = (float*)(ws + SZ);                           // scratch
    float*  bufB   = (float*)(ws + 2 * SZ);                       // scratch
    float*  quant1 = (float*)(ws + 3 * SZ);                       // persists
    __bf16* cbb1   = (__bf16*)(ws + 4 * SZ);
    __bf16* cbb2   = (__bf16*)(ws + 4 * SZ + (size_t)MMQ * KKQ * DDQ * 2);
    float*  c2a    = (float*)(ws + 4 * SZ + 2 * (size_t)MMQ * KKQ * DDQ * 2);
    float*  c2b    = c2a + MMQ * KKQ;

    const dim3 blk(256);
    const dim3 gGemm(512);   // 4096 wave-tiles / 8 waves per block
    const dim3 gQnt(128);    // N*M*8 pos-groups
    const dim3 gPrep(32);    // 8192 codebook rows / 256

    prep_codebook<<<gPrep, blk, 0, stream>>>(cb1, cbb1, c2a);
    prep_codebook<<<gPrep, blk, 0, stream>>>(cb2, cbb2, c2b);

    // ---- encode stage 1 ----
    gemm1x1_wmma<<<gGemm, blk, 0, stream>>>(bufZ, Wstage1, x, nullptr, nullptr);     // z1
    gemm1x1_wmma<<<gGemm, blk, 0, stream>>>(bufA, Wq1, bufZ, nullptr, nullptr);      // q1
    gemm1x1_wmma<<<gGemm, blk, 0, stream>>>(bufB, Wlat1, bufZ, nullptr, nullptr);    // latent
    quantize_wmma<<<gQnt, blk, 0, stream>>>(quant1, bufA, cb1, cbb1, c2a, tmp1, u1); // quant1

    // ---- encode stage 2 ----
    gemm1x1_wmma<<<gGemm, blk, 0, stream>>>(bufZ, Wstage2, bufB, nullptr, nullptr);  // z2
    gemm1x1_wmma<<<gGemm, blk, 0, stream>>>(bufA, Wq2, bufZ, nullptr, nullptr);      // q2
    quantize_wmma<<<gQnt, blk, 0, stream>>>(bufB, bufA, cb2, cbb2, c2b, tmp2, u2);   // quant2

    // ---- decode ----
    gemm1x1_wmma<<<gGemm, blk, 0, stream>>>(bufZ, Wdeq2, bufB, nullptr, nullptr);    // Wdeq2@quant2
    gemm1x1_wmma<<<gGemm, blk, 0, stream>>>(bufA, Wrest2, bufZ, nullptr, nullptr);   // r2
    gemm1x1_wmma<<<gGemm, blk, 0, stream>>>(bufB, Wdeq1, quant1, Wside1, bufA);      // fused sum
    gemm1x1_wmma<<<gGemm, blk, 0, stream>>>(out, Wrest1, bufB, nullptr, nullptr);    // r1
}
